// Model_25116968747264
// MI455X (gfx1250) — matrix operand order repaired and checked by reading, not compiled, not run on hardware
//
#include <hip/hip_runtime.h>
#include <hip/hip_bf16.h>

typedef __attribute__((ext_vector_type(16))) _Float16 v16h;
typedef __attribute__((ext_vector_type(8)))  _Float16 v8h;
typedef __attribute__((ext_vector_type(8)))  float    v8f;
typedef __attribute__((ext_vector_type(4))) unsigned int u32x4;
typedef __attribute__((ext_vector_type(8))) int i32x8;
typedef __attribute__((ext_vector_type(4))) int i32x4;

namespace cfg {
constexpr int E = 256, H = 256, T = 32, B = 64, S = 512;
constexpr int G = 4 * H;      // 1024 (gate width i,f,g,o)
constexpr int BS = B * S;     // 32768 GEMM rows
constexpr int WLD = 2 * H + 8;  // padded LDS row for Wlin (bank-conflict-free B frags)
}

// ---------------- CDNA5 feature probes ----------------
#if defined(__gfx1250__)
#define HAVE_ASYNC_LDS_ASM 1
#endif

#if defined(__has_builtin)
#if __has_builtin(__builtin_amdgcn_tensor_load_to_lds)
#define HAVE_TDM 1
#endif
#if __has_builtin(__builtin_amdgcn_s_wait_asynccnt)
#define WAIT_ASYNC(n) __builtin_amdgcn_s_wait_asynccnt((unsigned short)(n))
#else
#define WAIT_ASYNC(n) asm volatile("s_wait_asynccnt %0" ::"n"(n) : "memory")
#endif
#if __has_builtin(__builtin_amdgcn_s_wait_tensorcnt)
#define WAIT_TENSOR0() __builtin_amdgcn_s_wait_tensorcnt((unsigned short)0)
#else
#define WAIT_TENSOR0() asm volatile("s_wait_tensorcnt 0x0" ::: "memory")
#endif
#else
#define WAIT_ASYNC(n)
#define WAIT_TENSOR0()
#endif

#if defined(HAVE_TDM)
#if __has_include(<hip/amd_detail/amd_gfx1250_TDM.h>)
#define TDM_ARITY6 1
#endif
#endif

__device__ __forceinline__ float sigm(float x) { return 1.0f / (1.0f + __expf(-x)); }

// 16B async copy global->LDS (ASYNCcnt); inline asm per ISA 10.x VFLAT async form.
__device__ __forceinline__ void async_b128(const _Float16* g, _Float16* l) {
#ifdef HAVE_ASYNC_LDS_ASM
  unsigned lds = (unsigned)(unsigned long long)l;   // LDS aperture: low 32 bits
  unsigned long long ga = (unsigned long long)g;
  asm volatile("global_load_async_to_lds_b128 %0, %1, off"
               :: "v"(lds), "v"(ga) : "memory");
#else
  *(v8h*)l = *(const v8h*)g;
#endif
}

// TDM: 2D tile (tile_d0 4B-units x tile_d1 rows) global->LDS, optional LDS row padding.
// pad_interval_code: 0=2,1=4,...,7=256 dwords; pad_amount_code: n -> (n+1) dwords.
__device__ __forceinline__ void tdm_load_2d(const void* gptr, void* ldsptr,
                                            unsigned tile_d0, unsigned tile_d1,
                                            unsigned tensor_d0, unsigned tensor_d1,
                                            unsigned stride_d0, int pad_en,
                                            int pad_interval_code, int pad_amount_code) {
#ifdef HAVE_TDM
  unsigned long long ga = (unsigned long long)gptr;
  unsigned lds_off = (unsigned)(unsigned long long)ldsptr;  // LDS aperture: low 32 bits
  u32x4 g0;
  g0[0] = 1u;                                           // count=1, user descriptor
  g0[1] = lds_off;                                      // lds_addr
  g0[2] = (unsigned)(ga & 0xffffffffu);                 // global_addr lo
  g0[3] = (unsigned)((ga >> 32) & 0x01ffffffu) | (2u << 30);  // global_addr hi | type=2
  i32x8 g1;
  g1[0] = (int)((2u << 16) |                            // data_size = 4B
                ((unsigned)(pad_en ? 1 : 0) << 20) |
                ((unsigned)pad_interval_code << 22) |
                ((unsigned)pad_amount_code << 25));
  g1[1] = (int)((tensor_d0 & 0xffffu) << 16);           // tensor_dim0 lo16 @bits63:48
  g1[2] = (int)(((tensor_d0 >> 16) & 0xffffu) | ((tensor_d1 & 0xffffu) << 16));
  g1[3] = (int)(((tensor_d1 >> 16) & 0xffffu) | ((tile_d0 & 0xffffu) << 16));
  g1[4] = (int)(tile_d1 & 0xffffu);                     // tile_dim1 (tile_dim2 = 0)
  g1[5] = (int)stride_d0;                               // tensor_dim0_stride lo32
  g1[6] = 0;                                            // stride hi16 | dim1_stride lo16
  g1[7] = 0;
  i32x4 gz = {0, 0, 0, 0};
#ifdef TDM_ARITY6
  i32x8 gz8 = {0, 0, 0, 0, 0, 0, 0, 0};
  __builtin_amdgcn_tensor_load_to_lds(g0, g1, gz, gz, gz8, 0);
#else
  __builtin_amdgcn_tensor_load_to_lds(g0, g1, gz, gz, 0);
#endif
#else
  (void)gptr; (void)ldsptr; (void)tile_d0; (void)tile_d1; (void)tensor_d0;
  (void)tensor_d1; (void)stride_d0; (void)pad_en; (void)pad_interval_code;
  (void)pad_amount_code;
#endif
}

// ---------------- WMMA fragment helpers (CDNA5 wave32, ISA 7.12.2) ----------------
// A 16x32 f16: lane L holds row r=L&15; lanes 0-15 K={0..7,16..23}, lanes 16-31 K={8..15,24..31}
__device__ __forceinline__ v16h load_a_frag(const _Float16* base, int ld) {
  const int lane = threadIdx.x & 31;
  const int r = lane & 15, hi = lane >> 4;
  const _Float16* p = base + r * ld + hi * 8;
  v8h lo = *(const v8h*)(p);
  v8h hh = *(const v8h*)(p + 16);
  v16h a;
#pragma unroll
  for (int i = 0; i < 8; ++i) { a[i] = lo[i]; a[8 + i] = hh[i]; }
  return a;
}

// B 32x16 f16 (KxN), B[k,c] = W[n0+c, k0+k], W row-major [N,K]; wbase = W + n0*ld + k0.
__device__ __forceinline__ v16h load_b_frag(const _Float16* wbase, int ld) {
  const int lane = threadIdx.x & 31;
  const int c = lane & 15, hi = lane >> 4;
  const _Float16* p = wbase + c * ld + hi * 8;
  v8h lo = *(const v8h*)(p);
  v8h hh = *(const v8h*)(p + 16);
  v16h b;
#pragma unroll
  for (int i = 0; i < 8; ++i) { b[i] = lo[i]; b[8 + i] = hh[i]; }
  return b;
}

__device__ __forceinline__ v8f wmma_f16(v16h a, v16h b, v8f c) {
  return __builtin_amdgcn_wmma_f32_16x16x32_f16(false, a, false, b, (short)0, c, false, false);
}

// ---------------- small prep kernels ----------------
__global__ void k_f32_to_f16(const float* __restrict__ src, _Float16* __restrict__ dst, int n) {
  int i = blockIdx.x * blockDim.x + threadIdx.x;
  if (i < n) dst[i] = (_Float16)src[i];
}

__global__ void k_bias_sum(const float* __restrict__ a, const float* __restrict__ b,
                           float* __restrict__ out, int n) {
  int i = blockIdx.x * blockDim.x + threadIdx.x;
  if (i < n) out[i] = a[i] + b[i];
}

__global__ void k_embed(const int* __restrict__ input, const float* __restrict__ table,
                        _Float16* __restrict__ xh) {
  int idx = blockIdx.x * blockDim.x + threadIdx.x;   // BS*E/4 threads
  int row = idx >> 6;                                 // E/4 = 64 threads per row
  int e4 = (idx & 63) << 2;
  int tok = input[row];
  float4 v = *(const float4*)(table + tok * cfg::E + e4);
  _Float16* d = xh + row * cfg::E + e4;
  d[0] = (_Float16)v.x; d[1] = (_Float16)v.y; d[2] = (_Float16)v.z; d[3] = (_Float16)v.w;
}

// ---------------- input projection: xp = x @ Wih^T + (bih+bhh), f16 out ----------------
// grid: (N/64, M/128, dir), block 256 (8 waves). Each wave stages and consumes its OWN
// 16-row A sub-tile via async-to-LDS double buffering -> no block barriers at all.
// k-loop fully unrolled so accumulators stay in fixed registers (no phi-copy movs).
__global__ void k_xp_gemm(const _Float16* __restrict__ xh,
                          const _Float16* __restrict__ wih_f, const _Float16* __restrict__ wih_b,
                          const float* __restrict__ bsum_f, const float* __restrict__ bsum_b,
                          _Float16* __restrict__ xp_f, _Float16* __restrict__ xp_b) {
  const int dir = blockIdx.z;
  const _Float16* W = dir ? wih_b : wih_f;
  const float* bs = dir ? bsum_b : bsum_f;
  _Float16* xp = dir ? xp_b : xp_f;
  const int n0blk = blockIdx.x * 64, m0blk = blockIdx.y * 128;

  __shared__ _Float16 At[2][128 * 32];   // 2 x 8 KB A tiles
  const int w = threadIdx.x >> 5, lane = threadIdx.x & 31;
  const int srow = threadIdx.x >> 1;          // rows 16w..16w+15 staged by wave w
  const int sch = (threadIdx.x & 1) * 16;
  const _Float16* gsrc = xh + (m0blk + srow) * cfg::E + sch;
  _Float16* l0 = &At[0][srow * 32 + sch];
  _Float16* l1 = &At[1][srow * 32 + sch];

  async_b128(gsrc, l0);            // prologue: k-step 0 -> buf 0
  async_b128(gsrc + 8, l0 + 8);

  v8f acc[4] = {};
#pragma unroll
  for (int ks = 0; ks < 8; ++ks) {
    if (ks < 7) {                  // prefetch next k-step into alternate buffer
      _Float16* lb = ((ks + 1) & 1) ? l1 : l0;
      async_b128(gsrc + (ks + 1) * 32, lb);
      async_b128(gsrc + (ks + 1) * 32 + 8, lb + 8);
      WAIT_ASYNC(2);               // the 2 oldest (current buffer) complete
    } else {
      WAIT_ASYNC(0);
    }
    v16h a = load_a_frag(&At[ks & 1][(w * 16) * 32], 32);
#pragma unroll
    for (int j = 0; j < 4; ++j) {
      v16h b = load_b_frag(W + (n0blk + j * 16) * cfg::E + ks * 32, cfg::E);
      acc[j] = wmma_f16(a, b, acc[j]);
    }
  }

  const int hi = lane >> 4, c = lane & 15;
#pragma unroll
  for (int j = 0; j < 4; ++j) {
    int n = n0blk + j * 16 + c;
    float bv = bs[n];
#pragma unroll
    for (int i = 0; i < 8; ++i) {
      int m = m0blk + w * 16 + i + 8 * hi;
      xp[m * cfg::G + n] = (_Float16)(acc[j][i] + bv);
    }
  }
}

// ---------------- LSTM recurrence (one 32-wave block per direction) ----------------
__global__ void k_lstm(const _Float16* __restrict__ xp_f, const _Float16* __restrict__ xp_b,
                       const _Float16* __restrict__ whh_f, const _Float16* __restrict__ whh_b,
                       _Float16* __restrict__ hs_f, _Float16* __restrict__ hs_b) {
  const int dir = blockIdx.x;
  const _Float16* xp = dir ? xp_b : xp_f;
  const _Float16* Whh = dir ? whh_b : whh_f;
  _Float16* hs = dir ? hs_b : hs_f;

  __shared__ _Float16 hLds[cfg::B * cfg::H];   // 32 KB: h_{t-1} in f16
  const int w = threadIdx.x >> 5, lane = threadIdx.x & 31;
  const int mt = w & 3, nt = w >> 2;
  const int hi = lane >> 4, c = lane & 15;

  for (int i = threadIdx.x; i < cfg::B * cfg::H; i += 1024) hLds[i] = (_Float16)0.0f;
  float creg[2][8];
#pragma unroll
  for (int j = 0; j < 2; ++j)
#pragma unroll
    for (int i = 0; i < 8; ++i) creg[j][i] = 0.0f;
  __syncthreads();

  for (int step = 0; step < cfg::S; ++step) {
    const int t = dir ? (cfg::S - 1 - step) : step;

    // prefetch next step's xp gate slice while this step's GEMM runs
    if (step + 1 < cfg::S) {
      const int tn = dir ? (cfg::S - 2 - step) : (step + 1);
      const int pm = mt * 16 + (lane & 15);
      const int q0 = lane >> 4;
      const _Float16* pb = xp + (pm * cfg::S + tn) * cfg::G + nt * 32;
      __builtin_prefetch(pb + q0 * 256, 0, 1);
      __builtin_prefetch(pb + (q0 + 2) * 256, 0, 1);
    }

    v8f acc[4][2] = {};
    for (int ks = 0; ks < 8; ++ks) {
      v16h a = load_a_frag(hLds + (mt * 16) * cfg::H + ks * 32, cfg::H);
#pragma unroll
      for (int q = 0; q < 4; ++q)
#pragma unroll
        for (int j = 0; j < 2; ++j) {
          const int n0 = q * 256 + nt * 32 + j * 16;
          v16h b = load_b_frag(Whh + n0 * cfg::H + ks * 32, cfg::H);
          acc[q][j] = wmma_f16(a, b, acc[q][j]);
        }
    }
    __syncthreads();   // all reads of h_{t-1} complete

#pragma unroll
    for (int j = 0; j < 2; ++j)
#pragma unroll
      for (int i = 0; i < 8; ++i) {
        const int m = mt * 16 + i + 8 * hi;          // batch row
        const int hcol = nt * 32 + j * 16 + c;       // hidden col 0..255
        const int rowx = (m * cfg::S + t) * cfg::G;
        float iv = acc[0][j][i] + (float)xp[rowx + 0 * cfg::H + hcol];
        float fv = acc[1][j][i] + (float)xp[rowx + 1 * cfg::H + hcol];
        float gv = acc[2][j][i] + (float)xp[rowx + 2 * cfg::H + hcol];
        float ov = acc[3][j][i] + (float)xp[rowx + 3 * cfg::H + hcol];
        float cn = sigm(fv) * creg[j][i] + sigm(iv) * tanhf(gv);
        float hn = sigm(ov) * tanhf(cn);
        creg[j][i] = cn;
        _Float16 hh = (_Float16)hn;
        hLds[m * cfg::H + hcol] = hh;
        hs[(m * cfg::S + t) * cfg::H + hcol] = hh;
      }
    __syncthreads();   // h_t fully published before next step reads it
  }
}

// ---------------- emissions: em = [hf|hb] @ Wlin^T + blin ----------------
// Wlin staged into LDS by the Tensor Data Mover with hardware row padding
// (row stride 520 halves -> conflict-free ds_load_b128 B fragments).
__global__ void k_linear(const _Float16* __restrict__ hs_f, const _Float16* __restrict__ hs_b,
                         const _Float16* __restrict__ wlin, const float* __restrict__ blin,
                         float* __restrict__ em) {
  __shared__ _Float16 wl[cfg::T * cfg::WLD];   // 32 rows x (512+8) halves = 33280 B
  const int m0 = blockIdx.x * 128;
  const int w = threadIdx.x >> 5, lane = threadIdx.x & 31;
  const int r0 = m0 + w * 16;

#ifdef HAVE_TDM
  if (threadIdx.x < 32) {
    // 2D tile: 256 dwords/row x 32 rows, pad 4 dwords after every 256 dwords.
    tdm_load_2d(wlin, wl, /*tile_d0=*/256, /*tile_d1=*/32,
                /*tensor_d0=*/256, /*tensor_d1=*/32, /*stride_d0=*/256,
                /*pad_en=*/1, /*pad_interval_code=*/7, /*pad_amount_code=*/3);
    WAIT_TENSOR0();
  }
#else
  for (int i = threadIdx.x; i < cfg::T * (2 * cfg::H) / 8; i += 256) {
    int row = i >> 6;           // 64 chunks of 8 halves per row
    int ch = (i & 63) * 8;
    *(v8h*)(wl + row * cfg::WLD + ch) = *(const v8h*)(wlin + row * (2 * cfg::H) + ch);
  }
#endif
  __syncthreads();

  v8f acc[2] = {};
#pragma unroll
  for (int ks = 0; ks < 16; ++ks) {
    const _Float16* src = (ks < 8) ? hs_f : hs_b;
    const int k0 = (ks & 7) * 32;
    v16h a = load_a_frag(src + r0 * cfg::H + k0, cfg::H);
#pragma unroll
    for (int j = 0; j < 2; ++j) {
      v16h b = load_b_frag(wl + (j * 16) * cfg::WLD + ks * 32, cfg::WLD);
      acc[j] = wmma_f16(a, b, acc[j]);
    }
  }
  const int hi = lane >> 4, c = lane & 15;
#pragma unroll
  for (int j = 0; j < 2; ++j) {
    int tt = j * 16 + c;
    float bv = blin[tt];
#pragma unroll
    for (int i = 0; i < 8; ++i) {
      int r = r0 + i + 8 * hi;
      em[r * cfg::T + tt] = acc[j][i] + bv;
    }
  }
}

// ---------------- Viterbi: one wave32 per batch row; lane = tag ----------------
__global__ void k_viterbi(const float* __restrict__ em, const unsigned char* __restrict__ mask,
                          const float* __restrict__ start_trans, const float* __restrict__ end_trans,
                          const float* __restrict__ trans,
                          unsigned char* __restrict__ hist, float* __restrict__ out) {
  const int b = blockIdx.x;
  const int j = threadIdx.x;           // tag 0..31
  __shared__ float trl[cfg::T * cfg::T];
  __shared__ float sc[cfg::T];

  for (int i = j; i < cfg::T * cfg::T; i += 32) trl[i] = trans[i];
  float score = start_trans[j] + em[(b * cfg::S + 0) * cfg::T + j];
  __syncthreads();

  for (int t = 1; t < cfg::S; ++t) {
    sc[j] = score;
    __syncthreads();
    float e = em[(b * cfg::S + t) * cfg::T + j];
    float best = -3.4e38f; int bi = 0;
#pragma unroll 4
    for (int i = 0; i < cfg::T; ++i) {
      float v = sc[i] + trl[i * cfg::T + j];
      if (v > best) { best = v; bi = i; }
    }
    best += e;
    hist[((t - 1) * cfg::B + b) * cfg::T + j] = (unsigned char)bi;
    score = mask[b * cfg::S + t] ? best : score;
    __syncthreads();
  }

  score += end_trans[j];
  sc[j] = score;
  __threadfence();
  __syncthreads();
  if (j == 0) {
    float bb = sc[0]; int tag = 0;
    for (int i = 1; i < cfg::T; ++i) if (sc[i] > bb) { bb = sc[i]; tag = i; }
    out[b * cfg::S + (cfg::S - 1)] = (float)tag;
    for (int k = cfg::S - 2; k >= 0; --k) {
      int prev = hist[(k * cfg::B + b) * cfg::T + tag];
      tag = mask[b * cfg::S + (k + 1)] ? prev : tag;
      out[b * cfg::S + k] = (float)tag;
    }
  }
}

// ---------------- host ----------------
extern "C" void kernel_launch(void* const* d_in, const int* in_sizes, int n_in,
                              void* d_out, int out_size, void* d_ws, size_t ws_size,
                              hipStream_t stream) {
  using namespace cfg;
  const int*            input = (const int*)d_in[0];
  const unsigned char*  mask  = (const unsigned char*)d_in[1];
  const float* table = (const float*)d_in[2];
  const float* Wih_f = (const float*)d_in[3];
  const float* Whh_f = (const float*)d_in[4];
  const float* bih_f = (const float*)d_in[5];
  const float* bhh_f = (const float*)d_in[6];
  const float* Wih_b = (const float*)d_in[7];
  const float* Whh_b = (const float*)d_in[8];
  const float* bih_b = (const float*)d_in[9];
  const float* bhh_b = (const float*)d_in[10];
  const float* Wlin  = (const float*)d_in[11];
  const float* blin  = (const float*)d_in[12];
  const float* st    = (const float*)d_in[13];
  const float* et    = (const float*)d_in[14];
  const float* tr    = (const float*)d_in[15];
  float* out = (float*)d_out;

  char* ws = (char*)d_ws;
  size_t off = 0;
  auto alloc = [&](size_t bytes) { void* p = ws + off; off += (bytes + 255) & ~(size_t)255; return p; };

  _Float16* xh     = (_Float16*)alloc((size_t)BS * E * 2);
  _Float16* wihf_h = (_Float16*)alloc((size_t)G * E * 2);
  _Float16* whhf_h = (_Float16*)alloc((size_t)G * H * 2);
  _Float16* wihb_h = (_Float16*)alloc((size_t)G * E * 2);
  _Float16* whhb_h = (_Float16*)alloc((size_t)G * H * 2);
  _Float16* wlin_h = (_Float16*)alloc((size_t)T * 2 * H * 2);
  float*    bsum_f = (float*)alloc((size_t)G * 4);
  float*    bsum_b = (float*)alloc((size_t)G * 4);
  _Float16* xp_f   = (_Float16*)alloc((size_t)BS * G * 2);
  _Float16* xp_b   = (_Float16*)alloc((size_t)BS * G * 2);
  _Float16* hs_f   = (_Float16*)alloc((size_t)BS * H * 2);
  _Float16* hs_b   = (_Float16*)alloc((size_t)BS * H * 2);
  float*    em     = (float*)alloc((size_t)BS * T * 4);
  unsigned char* hist = (unsigned char*)alloc((size_t)(S - 1) * B * T);

  k_f32_to_f16<<<(G * E + 255) / 256, 256, 0, stream>>>(Wih_f, wihf_h, G * E);
  k_f32_to_f16<<<(G * H + 255) / 256, 256, 0, stream>>>(Whh_f, whhf_h, G * H);
  k_f32_to_f16<<<(G * E + 255) / 256, 256, 0, stream>>>(Wih_b, wihb_h, G * E);
  k_f32_to_f16<<<(G * H + 255) / 256, 256, 0, stream>>>(Whh_b, whhb_h, G * H);
  k_f32_to_f16<<<(T * 2 * H + 255) / 256, 256, 0, stream>>>(Wlin, wlin_h, T * 2 * H);
  k_bias_sum<<<(G + 255) / 256, 256, 0, stream>>>(bih_f, bhh_f, bsum_f, G);
  k_bias_sum<<<(G + 255) / 256, 256, 0, stream>>>(bih_b, bhh_b, bsum_b, G);

  k_embed<<<(BS * E / 4 + 255) / 256, 256, 0, stream>>>(input, table, xh);

  k_xp_gemm<<<dim3(G / 64, BS / 128, 2), 256, 0, stream>>>(xh, wihf_h, wihb_h,
                                                           bsum_f, bsum_b, xp_f, xp_b);

  k_lstm<<<2, 1024, 0, stream>>>(xp_f, xp_b, whhf_h, whhb_h, hs_f, hs_b);

  k_linear<<<BS / 128, 256, 0, stream>>>(hs_f, hs_b, wlin_h, blin, em);

  k_viterbi<<<B, 32, 0, stream>>>(em, mask, st, et, tr, hist, out);

  (void)in_sizes; (void)n_in; (void)out_size; (void)ws_size;
}
